// SwinBlock3D_28750511080021
// MI455X (gfx1250) — matrix operand order repaired and checked by reading, not compiled, not run on hardware
//
#include <hip/hip_runtime.h>

// ---------------- types & helpers ----------------
typedef __attribute__((ext_vector_type(16))) __bf16 v16bf;
typedef __attribute__((ext_vector_type(8)))  float  v8f;
typedef __attribute__((ext_vector_type(8)))  unsigned short u16x8;
typedef __attribute__((ext_vector_type(16))) unsigned short u16x16;

#define DIM 96
#define HEADS 3
#define HD 32
#define WV 64
#define SHIFT 2
#define MLPD 384

// padded LDS row strides (ushorts / floats) - chosen so row base stays 16B
// aligned and lane bank offsets avoid 16B-window overlap within a half-wave
#define XLP 104   // xln / obuf rows (96 data)
#define QKP 40    // qbuf / kbuf rows (32 data)
#define VTP 72    // vtbf rows (64 data)
#define ABP 72    // abuf rows (64 data)
#define SBP 65    // sbuf rows, fp32 (64 data)
#define HBP 392   // hbuf rows (384 data)

__device__ __forceinline__ unsigned short f2bf(float f) {
  unsigned int u = __float_as_uint(f);
  u += 0x7FFFu + ((u >> 16) & 1u);            // round-to-nearest-even
  return (unsigned short)(u >> 16);
}

// A-fragment (16x32 bf16): lane<16 -> M=lane, K=0..15; lane>=16 -> K=16..31.
__device__ __forceinline__ v16bf afrag(const unsigned short* row, int kstep, int half) {
  const u16x8 lo = *(const u16x8*)(row + kstep * 32 + half * 16);
  const u16x8 hi = *(const u16x8*)(row + kstep * 32 + half * 16 + 8);
  u16x16 u;
#pragma unroll
  for (int i = 0; i < 8; ++i) { u[i] = lo[i]; u[i + 8] = hi[i]; }
  return __builtin_bit_cast(v16bf, u);
}

// B-fragment (32x16 bf16): lane -> column, lanes 0-15 hold K=0..15, 16-31 K=16..31.
__device__ __forceinline__ v16bf bfrag(const unsigned short* p) {
  return __builtin_bit_cast(v16bf, *(const u16x16*)p);
}

__device__ __forceinline__ v8f wmma_bf16(v16bf a, v16bf b, v8f c) {
  return __builtin_amdgcn_wmma_f32_16x16x32_bf16(false, a, false, b, (short)0, c, false, false);
}

// token t (0..63) in window (wx,wy,wz) -> flat token index in ORIGINAL x
__device__ __forceinline__ int gtok(int wx, int wy, int wz, int t) {
  int tx = t >> 4, ty = (t >> 2) & 3, tz = t & 3;
  int X = (wx * 4 + tx + SHIFT) & 63;
  int Y = (wy * 4 + ty + SHIFT) & 63;
  int Z = (wz * 4 + tz + SHIFT) & 63;
  return (X << 12) | (Y << 6) | Z;
}

__device__ __forceinline__ int region(int q) { return q < 60 ? 0 : (q < 62 ? 1 : 2); }

// ---------------- weight pre-swizzle: fp32 [K][N] -> bf16 fragment-contiguous ----------------
__global__ __launch_bounds__(256) void swizzleW(const float* __restrict__ src,
                                                unsigned short* __restrict__ dst,
                                                int N, int K) {
  int d = blockIdx.x * 256 + threadIdx.x;
  if (d >= N * K) return;
  int tile = d >> 9, lane = (d >> 4) & 31, e = d & 15;
  int ksteps = K >> 5;
  int ntile = tile / ksteps, kstep = tile % ksteps;
  int n = ntile * 16 + (lane & 15);
  int k = kstep * 32 + (lane >> 4) * 16 + e;
  dst[d] = f2bf(src[k * N + n]);
}

// ---------------- kernel 1: LN1 + shifted-window attention + proj + residual ----------------
__global__ __launch_bounds__(256) void swin_attn_kernel(
    const float* __restrict__ x,
    const unsigned short* __restrict__ wqkv,   // swizzled [18 ntiles][3 ksteps][512]
    const unsigned short* __restrict__ wout,   // swizzled [6][3][512]
    const float* __restrict__ b_out,
    const float* __restrict__ pos_tab,
    const float* __restrict__ ln_g, const float* __restrict__ ln_b,
    float* __restrict__ out) {
  __shared__ __align__(16) float          xwin[WV * DIM];     // 24 KB residual + LN src
  __shared__ __align__(16) unsigned short xln [WV * XLP];     // 13 KB
  __shared__ __align__(16) unsigned short qbuf[WV * QKP];     // 5 KB  [tok][dim]
  __shared__ __align__(16) unsigned short kbuf[WV * QKP];     // 5 KB  [tok][dim]
  __shared__ __align__(16) unsigned short vtbf[HD * VTP];     // 4.5 KB [dim][tok]
  __shared__ __align__(16) float          sbuf[WV * SBP];     // 16.25 KB scores
  __shared__ __align__(16) unsigned short abuf[WV * ABP];     // 9 KB attn bf16
  __shared__ __align__(16) unsigned short obuf[WV * XLP];     // 13 KB per-head concat
  __shared__               float          ptab[343];
  __shared__               float          gbuf[DIM], bbuf[DIM];

  const int tid = threadIdx.x;
  const int wave = tid >> 5, lane = tid & 31, half = lane >> 4, lr = lane & 15;
  const int w = blockIdx.x;
  const int wx = w >> 8, wy = (w >> 4) & 15, wz = w & 15;

  __builtin_prefetch(wqkv, 0, 3);
  __builtin_prefetch(wout, 0, 3);

  // ---- gather shifted window (float4 loads) + constants ----
  for (int i = tid; i < WV * 24; i += 256) {
    int t = i / 24, c4 = i % 24;
    const float4 v = ((const float4*)(x + (size_t)gtok(wx, wy, wz, t) * DIM))[c4];
    *(float4*)&xwin[t * DIM + c4 * 4] = v;
  }
  for (int i = tid; i < 343; i += 256) ptab[i] = pos_tab[i];
  for (int i = tid; i < DIM; i += 256) { gbuf[i] = ln_g[i]; bbuf[i] = ln_b[i]; }
  __syncthreads();

  // ---- LayerNorm1 -> bf16 (4 lanes per row, shfl reduction) ----
  {
    const int i = tid >> 2, qd = tid & 3;
    const float* row = &xwin[i * DIM];
    float s = 0.f, sq = 0.f;
#pragma unroll
    for (int c = qd * 24; c < qd * 24 + 24; ++c) { float v = row[c]; s += v; sq += v * v; }
    s  += __shfl_xor(s, 1);  s  += __shfl_xor(s, 2);
    sq += __shfl_xor(sq, 1); sq += __shfl_xor(sq, 2);
    float m = s * (1.f / DIM);
    float rs = rsqrtf(sq * (1.f / DIM) - m * m + 1e-5f);
#pragma unroll
    for (int c = qd * 24; c < qd * 24 + 24; ++c)
      xln[i * XLP + c] = f2bf((row[c] - m) * rs * gbuf[c] + bbuf[c]);
  }
  __syncthreads();

  // ---- per-head attention ----
  for (int h = 0; h < HEADS; ++h) {
    // QKV GEMM slice for head h: 6 column-tiles x 4 row-tiles = 24 tiles, 3 per wave
#pragma unroll
    for (int s = 0; s < 3; ++s) {
      int tt = wave * 3 + s;
      int colgrp = tt >> 2, mtile = tt & 3;
      int ntile = (colgrp < 2) ? (2 * h + colgrp)
                : (colgrp < 4) ? (6 + 2 * h + (colgrp - 2))
                               : (12 + 2 * h + (colgrp - 4));
      v8f acc = {};
#pragma unroll
      for (int ks = 0; ks < 3; ++ks) {
        v16bf a = afrag(&xln[(mtile * 16 + lr) * XLP], ks, half);
        v16bf b = bfrag(wqkv + ((ntile * 3 + ks) << 9) + lane * 16);
        acc = wmma_bf16(a, b, acc);
      }
      if (colgrp < 2) {
        int dim = colgrp * 16 + lr;
#pragma unroll
        for (int r = 0; r < 8; ++r) qbuf[(mtile * 16 + half * 8 + r) * QKP + dim] = f2bf(acc[r]);
      } else if (colgrp < 4) {
        int dim = (colgrp - 2) * 16 + lr;
#pragma unroll
        for (int r = 0; r < 8; ++r) kbuf[(mtile * 16 + half * 8 + r) * QKP + dim] = f2bf(acc[r]);
      } else {
        int dim = (colgrp - 4) * 16 + lr;
        u16x8 p;
#pragma unroll
        for (int r = 0; r < 8; ++r) p[r] = f2bf(acc[r]);
        *(u16x8*)&vtbf[dim * VTP + mtile * 16 + half * 8] = p;   // V stored transposed
      }
    }
    __syncthreads();

    // S = Q K^T : 16 tiles, 2 per wave (K=32, one WMMA)
#pragma unroll
    for (int s = 0; s < 2; ++s) {
      int tt = wave * 2 + s;
      int itile = tt >> 2, jtile = tt & 3;
      v16bf a = afrag(&qbuf[(itile * 16 + lr) * QKP], 0, half);
      v16bf b = bfrag(&kbuf[(jtile * 16 + lr) * QKP + half * 16]);
      v8f acc = {};
      acc = wmma_bf16(a, b, acc);
#pragma unroll
      for (int r = 0; r < 8; ++r)
        sbuf[(itile * 16 + half * 8 + r) * SBP + jtile * 16 + lr] = acc[r];
    }
    __syncthreads();

    // softmax (scale + rel-pos bias + shifted-window mask), 4 lanes per row
    {
      const int i = tid >> 2, qd = tid & 3, j0 = qd * 16;
      const int txi = i >> 4, tyi = (i >> 2) & 3, tzi = i & 3;
      const int labi = region(wx * 4 + txi) * 9 + region(wy * 4 + tyi) * 3 + region(wz * 4 + tzi);
      const float scale = 0.17677669529663687f;   // 1/sqrt(32)
      float mx = -1e30f;
#pragma unroll
      for (int jj = 0; jj < 16; ++jj) {
        int j = j0 + jj;
        int txj = j >> 4, tyj = (j >> 2) & 3, tzj = j & 3;
        int labj = region(wx * 4 + txj) * 9 + region(wy * 4 + tyj) * 3 + region(wz * 4 + tzj);
        float bias = ptab[((txi - txj + 3) * 7 + (tyi - tyj + 3)) * 7 + (tzi - tzj + 3)];
        float v = sbuf[i * SBP + j] * scale + bias + (labi != labj ? -1e9f : 0.f);
        sbuf[i * SBP + j] = v;
        mx = fmaxf(mx, v);
      }
      mx = fmaxf(mx, __shfl_xor(mx, 1));
      mx = fmaxf(mx, __shfl_xor(mx, 2));
      float sum = 0.f;
#pragma unroll
      for (int jj = 0; jj < 16; ++jj) {
        float e = __expf(sbuf[i * SBP + j0 + jj] - mx);
        sbuf[i * SBP + j0 + jj] = e;
        sum += e;
      }
      sum += __shfl_xor(sum, 1);
      sum += __shfl_xor(sum, 2);
      float rinv = 1.f / sum;
#pragma unroll
      for (int jj = 0; jj < 16; ++jj)
        abuf[i * ABP + j0 + jj] = f2bf(sbuf[i * SBP + j0 + jj] * rinv);
    }
    __syncthreads();

    // O_h = attn @ V : 8 tiles, 1 per wave (K=64 -> 2 WMMAs)
    {
      int mtile = wave >> 1, ntile = wave & 1;
      v8f acc = {};
#pragma unroll
      for (int ks = 0; ks < 2; ++ks) {
        v16bf a = afrag(&abuf[(mtile * 16 + lr) * ABP], ks, half);
        v16bf b = bfrag(&vtbf[(ntile * 16 + lr) * VTP + ks * 32 + half * 16]);
        acc = wmma_bf16(a, b, acc);
      }
      int col = h * HD + ntile * 16 + lr;
#pragma unroll
      for (int r = 0; r < 8; ++r) obuf[(mtile * 16 + half * 8 + r) * XLP + col] = f2bf(acc[r]);
    }
    __syncthreads();
  }

  // ---- out projection + bias + residual, scatter back (shift folds away) ----
#pragma unroll
  for (int s = 0; s < 3; ++s) {
    int tt = wave * 3 + s;
    int mtile = tt & 3, ntile = tt >> 2;
    v8f acc = {};
#pragma unroll
    for (int ks = 0; ks < 3; ++ks) {
      v16bf a = afrag(&obuf[(mtile * 16 + lr) * XLP], ks, half);
      v16bf b = bfrag(wout + ((ntile * 3 + ks) << 9) + lane * 16);
      acc = wmma_bf16(a, b, acc);
    }
    int n = ntile * 16 + lr;
    float bo = b_out[n];
#pragma unroll
    for (int r = 0; r < 8; ++r) {
      int m = mtile * 16 + half * 8 + r;
      out[(size_t)gtok(wx, wy, wz, m) * DIM + n] = xwin[m * DIM + n] + acc[r] + bo;
    }
  }
}

// ---------------- kernel 2: LN2 + MLP (W1 -> GELU -> W2) + residual (RMW on d_out) ----------------
__global__ __launch_bounds__(256) void mlp_kernel(
    const float* __restrict__ ln_g, const float* __restrict__ ln_b,
    const unsigned short* __restrict__ w1s,   // swizzled [24][3][512]
    const float* __restrict__ b1,
    const unsigned short* __restrict__ w2s,   // swizzled [6][12][512]
    const float* __restrict__ b2,
    float* __restrict__ out) {
  __shared__ __align__(16) float          xwin[WV * DIM];    // 24 KB
  __shared__ __align__(16) unsigned short xln [WV * XLP];    // 13 KB
  __shared__ __align__(16) unsigned short hbuf[WV * HBP];    // 49 KB
  __shared__               float          gbuf[DIM], bbuf[DIM];

  const int tid = threadIdx.x;
  const int wave = tid >> 5, lane = tid & 31, half = lane >> 4, lr = lane & 15;
  const size_t base = (size_t)blockIdx.x * WV * DIM;

  __builtin_prefetch(w1s, 0, 3);
  __builtin_prefetch(w2s, 0, 3);

  for (int i = tid; i < WV * 24; i += 256)
    *(float4*)&xwin[i * 4] = ((const float4*)(out + base))[i];
  for (int i = tid; i < DIM; i += 256) { gbuf[i] = ln_g[i]; bbuf[i] = ln_b[i]; }
  __syncthreads();

  // LayerNorm2 (4 lanes per row)
  {
    const int i = tid >> 2, qd = tid & 3;
    const float* row = &xwin[i * DIM];
    float s = 0.f, sq = 0.f;
#pragma unroll
    for (int c = qd * 24; c < qd * 24 + 24; ++c) { float v = row[c]; s += v; sq += v * v; }
    s  += __shfl_xor(s, 1);  s  += __shfl_xor(s, 2);
    sq += __shfl_xor(sq, 1); sq += __shfl_xor(sq, 2);
    float m = s * (1.f / DIM);
    float rs = rsqrtf(sq * (1.f / DIM) - m * m + 1e-5f);
#pragma unroll
    for (int c = qd * 24; c < qd * 24 + 24; ++c)
      xln[i * XLP + c] = f2bf((row[c] - m) * rs * gbuf[c] + bbuf[c]);
  }
  __syncthreads();

  // GEMM1: [64x96] @ [96x384] -> GELU -> hbuf ; 96 tiles, 12 per wave
#pragma unroll
  for (int s = 0; s < 12; ++s) {
    int tt = wave * 12 + s;
    int ntile = tt >> 2, mtile = tt & 3;
    v8f acc = {};
#pragma unroll
    for (int ks = 0; ks < 3; ++ks) {
      v16bf a = afrag(&xln[(mtile * 16 + lr) * XLP], ks, half);
      v16bf b = bfrag(w1s + ((ntile * 3 + ks) << 9) + lane * 16);
      acc = wmma_bf16(a, b, acc);
    }
    int n = ntile * 16 + lr;
    float bb = b1[n];
#pragma unroll
    for (int r = 0; r < 8; ++r) {
      float v = acc[r] + bb;
      float g = 0.5f * v * (1.f + erff(v * 0.70710678118654752f));   // exact GELU
      hbuf[(mtile * 16 + half * 8 + r) * HBP + n] = f2bf(g);
    }
  }
  __syncthreads();

  // GEMM2: [64x384] @ [384x96] + b2 + residual ; 24 tiles, 3 per wave
#pragma unroll
  for (int s = 0; s < 3; ++s) {
    int tt = wave * 3 + s;
    int mtile = tt & 3, ntile = tt >> 2;
    v8f acc = {};
#pragma unroll
    for (int ks = 0; ks < 12; ++ks) {
      v16bf a = afrag(&hbuf[(mtile * 16 + lr) * HBP], ks, half);
      v16bf b = bfrag(w2s + ((ntile * 12 + ks) << 9) + lane * 16);
      acc = wmma_bf16(a, b, acc);
    }
    int n = ntile * 16 + lr;
    float bb = b2[n];
#pragma unroll
    for (int r = 0; r < 8; ++r) {
      int m = mtile * 16 + half * 8 + r;
      out[base + m * DIM + n] = xwin[m * DIM + n] + acc[r] + bb;
    }
  }
}

// ---------------- launch ----------------
extern "C" void kernel_launch(void* const* d_in, const int* in_sizes, int n_in,
                              void* d_out, int out_size, void* d_ws, size_t ws_size,
                              hipStream_t stream) {
  (void)in_sizes; (void)n_in; (void)out_size; (void)ws_size;
  const float* x      = (const float*)d_in[0];
  const float* W_qkv  = (const float*)d_in[1];
  const float* W_out  = (const float*)d_in[2];
  const float* b_out  = (const float*)d_in[3];
  const float* ptab   = (const float*)d_in[4];
  const float* ln1_g  = (const float*)d_in[5];
  const float* ln1_b  = (const float*)d_in[6];
  const float* ln2_g  = (const float*)d_in[7];
  const float* ln2_b  = (const float*)d_in[8];
  const float* W1     = (const float*)d_in[9];
  const float* b1     = (const float*)d_in[10];
  const float* W2     = (const float*)d_in[11];
  const float* b2     = (const float*)d_in[12];
  float* out = (float*)d_out;

  unsigned short* ws = (unsigned short*)d_ws;
  unsigned short* wqkv_s = ws;                 // 96*288  = 27648
  unsigned short* wout_s = ws + 27648;         // 96*96   =  9216
  unsigned short* w1_s   = ws + 36864;         // 96*384  = 36864
  unsigned short* w2_s   = ws + 73728;         // 384*96  = 36864  (total 221 KB)

  swizzleW<<<(27648 + 255) / 256, 256, 0, stream>>>(W_qkv, wqkv_s, 288, 96);
  swizzleW<<<( 9216 + 255) / 256, 256, 0, stream>>>(W_out, wout_s,  96, 96);
  swizzleW<<<(36864 + 255) / 256, 256, 0, stream>>>(W1,    w1_s,   384, 96);
  swizzleW<<<(36864 + 255) / 256, 256, 0, stream>>>(W2,    w2_s,    96, 384);

  swin_attn_kernel<<<4096, 256, 0, stream>>>(x, wqkv_s, wout_s, b_out, ptab,
                                             ln1_g, ln1_b, out);
  mlp_kernel<<<4096, 256, 0, stream>>>(ln2_g, ln2_b, w1_s, b1, w2_s, b2, out);
}
